// ComplexAttention_27977416966203
// MI455X (gfx1250) — matrix operand order repaired and checked by reading, not compiled, not run on hardware
//
#include <hip/hip_runtime.h>

// ---------------------------------------------------------------------------
// ComplexAttention on MI455X (gfx1250, wave32, WMMA)
//   B=8, S=2048, D=512, KD=64, WIDTH=128  -> banded causal attention
//
// bf16 operands produced once in workspace (qi pre-negated, V transposed);
// weights staged in LDS in WMMA B-fragment order so the GEMM inner loop is
// 2x ds_load_b128 + v_wmma per fragment.
// ---------------------------------------------------------------------------

#define BATCH 8
#define SEQ   2048
#define DIM   512
#define KD    64
#define QSCALE 0.125f   // 1/sqrt(64)
#define NEGBIG 1.0e9f

typedef __attribute__((ext_vector_type(16))) __bf16 v16bf;
typedef __attribute__((ext_vector_type(8)))  float  v8f;
typedef __attribute__((ext_vector_type(2)))  float  v2f;
typedef __attribute__((ext_vector_type(2)))  __bf16 v2bf;

union VB { v16bf v; unsigned u[8]; };

static __device__ __forceinline__ unsigned pack2(float lo, float hi) {
    v2f f = {lo, hi};
    union { v2bf b; unsigned u; } r;
    r.b = __builtin_convertvector(f, v2bf);   // v_cvt_pk_bf16_f32
    return r.u;
}
static __device__ __forceinline__ v8f wmma_bf16(v16bf a, v16bf b, v8f c) {
    return __builtin_amdgcn_wmma_f32_16x16x32_bf16(
        false, a, false, b, (short)0, c, false, false);
}

// ---------------------------------------------------------------------------
// Kernel 1: fused projections.  grid=(256, 5), block=128 (4 waves).
//   p=0: qr=(x@Wqr+b)*s+pos      -> bf16 [row][64]
//   p=1: qi likewise, NEGATED    -> bf16 [row][64]
//   p=2/3: kr/ki = x@W+b+pos     -> bf16 [row][64]
//   p=4: v = x@Wv+b              -> bf16 TRANSPOSED [b][d][s]
// W staged in LDS in B-fragment order: [kc][nt][g][lane] x uint4.
// ---------------------------------------------------------------------------
__global__ __launch_bounds__(128) void proj_kernel(
    const float* __restrict__ x,
    const float* __restrict__ Wqr, const float* __restrict__ bqr,
    const float* __restrict__ Wqi, const float* __restrict__ bqi,
    const float* __restrict__ Wkr, const float* __restrict__ bkr,
    const float* __restrict__ Wki, const float* __restrict__ bki,
    const float* __restrict__ Wv,  const float* __restrict__ bv,
    const float* __restrict__ pqr, const float* __restrict__ pqi,
    const float* __restrict__ pkr, const float* __restrict__ pki,
    __bf16* __restrict__ oqr, __bf16* __restrict__ oqiN,
    __bf16* __restrict__ okr, __bf16* __restrict__ oki,
    __bf16* __restrict__ ovT)
{
    // fragment-order weights: ((kc*4 + nt)*2 + g)*32 + lane -> uint4 (64 KB)
    __shared__ uint4 Wlds4[16 * 4 * 2 * 32];

    const int p = blockIdx.y;
    const float* W;  const float* bias; const float* pos; __bf16* out;
    switch (p) {
        case 0: W = Wqr; bias = bqr; pos = pqr; out = oqr;  break;
        case 1: W = Wqi; bias = bqi; pos = pqi; out = oqiN; break;
        case 2: W = Wkr; bias = bkr; pos = pkr; out = okr;  break;
        case 3: W = Wki; bias = bki; pos = pki; out = oki;  break;
        default: W = Wv; bias = bv;  pos = (const float*)0; out = (__bf16*)0; break;
    }

    const int tid = threadIdx.x;
    // Stage W: coalesced global reads (n fastest), scatter to fragment order.
    // Element (k=2*k2, n) -> fragment (kc=k>>5, nt=n>>4, g=j>>2, lane=hh*16+n%16),
    // dword j&3, where hh=(k>>4)&1, j=(k>>1)&7.
    {
        unsigned* Wd = (unsigned*)Wlds4;
        for (int idx = tid; idx < (DIM / 2) * KD; idx += 128) {
            int n  = idx & 63;
            int k  = (idx >> 6) << 1;
            unsigned val = pack2(W[k * KD + n], W[(k + 1) * KD + n]);
            int kc = k >> 5, hh2 = (k >> 4) & 1, j = (k >> 1) & 7;
            int q = ((kc * 4 + (n >> 4)) * 2 + (j >> 2)) * 32 + hh2 * 16 + (n & 15);
            Wd[q * 4 + (j & 3)] = val;
        }
    }
    __syncthreads();

    const int wave = tid >> 5, lane = tid & 31;
    const int hh   = lane >> 4;
    const int n15  = lane & 15;
    const int rowbase = blockIdx.x * 64 + wave * 16;
    const float* xrow = x + (size_t)(rowbase + n15) * DIM;   // A: lane%16 = M

    v8f acc[4];
#pragma unroll
    for (int nt = 0; nt < 4; ++nt)
#pragma unroll
        for (int r = 0; r < 8; ++r) acc[nt][r] = 0.0f;

    for (int kc = 0; kc < 16; ++kc) {
        const int kb = kc * 32;
        if (kc < 15)
            __builtin_prefetch(xrow + kb + 32, 0, 1);   // global_prefetch_b8

        // A operand: 4x global b128 + packed bf16 cvt
        VB a;
        float4 xa = *(const float4*)(xrow + kb + hh * 16);
        float4 xb = *(const float4*)(xrow + kb + hh * 16 + 4);
        float4 xc = *(const float4*)(xrow + kb + hh * 16 + 8);
        float4 xd = *(const float4*)(xrow + kb + hh * 16 + 12);
        a.u[0] = pack2(xa.x, xa.y); a.u[1] = pack2(xa.z, xa.w);
        a.u[2] = pack2(xb.x, xb.y); a.u[3] = pack2(xb.z, xb.w);
        a.u[4] = pack2(xc.x, xc.y); a.u[5] = pack2(xc.z, xc.w);
        a.u[6] = pack2(xd.x, xd.y); a.u[7] = pack2(xd.z, xd.w);

#pragma unroll
        for (int nt = 0; nt < 4; ++nt) {
            // B operand: two ds_load_b128, lane-major 16B stride
            const uint4* f = Wlds4 + ((kc * 4 + nt) * 2) * 32 + lane;
            uint4 b0 = f[0];
            uint4 b1 = f[32];
            VB bm;
            bm.u[0] = b0.x; bm.u[1] = b0.y; bm.u[2] = b0.z; bm.u[3] = b0.w;
            bm.u[4] = b1.x; bm.u[5] = b1.y; bm.u[6] = b1.z; bm.u[7] = b1.w;
            acc[nt] = wmma_bf16(a.v, bm.v, acc[nt]);
        }
    }

    // Epilogue, C-layout element (m = r + 8*hh, n = nt*16 + n15).
    if (p == 4) {
        // V transposed: vT[b][d][s], 8 consecutive s per lane -> one b128.
        const int bidx = rowbase >> 11;
        const int s0 = (rowbase & (SEQ - 1)) + (hh << 3);
#pragma unroll
        for (int nt = 0; nt < 4; ++nt) {
            int n = nt * 16 + n15;
            float bn = bias[n];
            uint4 q;
            q.x = pack2(acc[nt][0] + bn, acc[nt][1] + bn);
            q.y = pack2(acc[nt][2] + bn, acc[nt][3] + bn);
            q.z = pack2(acc[nt][4] + bn, acc[nt][5] + bn);
            q.w = pack2(acc[nt][6] + bn, acc[nt][7] + bn);
            *(uint4*)(ovT + ((size_t)(bidx * KD + n)) * SEQ + s0) = q;
        }
    } else {
#pragma unroll
        for (int nt = 0; nt < 4; ++nt) {
            int n = nt * 16 + n15;
            float bn = bias[n];
#pragma unroll
            for (int r = 0; r < 8; ++r) {
                int row = rowbase + r + (hh << 3);
                float o = acc[nt][r] + bn;
                if (p <= 1) o = o * QSCALE + pos[(row & (SEQ - 1)) * KD + n];
                else        o = o + pos[(row & (SEQ - 1)) * KD + n];
                if (p == 1) o = -o;                      // pre-negate qi
                out[(size_t)row * KD + n] = (__bf16)o;
            }
        }
    }
}

// ---------------------------------------------------------------------------
// Kernel 2: banded causal attention.  grid=256, block=128 (4 waves).
// One wave per (b, 16-query tile); key tiles kt in [qt-8, qt] (<=9 live).
// ---------------------------------------------------------------------------
#define SCS 164   // padded row stride: conflict-free mod 64 banks, 16B-aligned

__global__ __launch_bounds__(128) void attn_kernel(
    const __bf16* __restrict__ qr, const __bf16* __restrict__ qiN,
    const __bf16* __restrict__ kr, const __bf16* __restrict__ ki,
    const __bf16* __restrict__ vT, const float* __restrict__ temp_p,
    float* __restrict__ out)
{
    __shared__ __align__(16) float          sc[4][16 * SCS];  // scores f32
    __shared__ __align__(16) unsigned short pb[4][16 * SCS];  // probs bf16

    const int tid = threadIdx.x, wave = tid >> 5, lane = tid & 31;
    const int gid = blockIdx.x * 4 + wave;
    const int b = gid >> 7, qt = gid & 127;
    const int hh = lane >> 4, n15 = lane & 15;
    const float temp = temp_p[0];

    float*          scw = sc[wave];
    unsigned short* pbw = pb[wave];

    // ---- load Q tiles (A layout) as raw bf16 dwords: 2x b128 per chunk ----
    const __bf16* qrb = qr  + (size_t)(b * SEQ + qt * 16 + n15) * KD;
    const __bf16* qib = qiN + (size_t)(b * SEQ + qt * 16 + n15) * KD;
    VB qrA[2], qiA[2];
#pragma unroll
    for (int c = 0; c < 2; ++c) {
        int kb0 = c * 32 + hh * 16;
        uint4 qa = *(const uint4*)(qrb + kb0);
        uint4 qb = *(const uint4*)(qrb + kb0 + 8);
        qrA[c].u[0] = qa.x; qrA[c].u[1] = qa.y; qrA[c].u[2] = qa.z; qrA[c].u[3] = qa.w;
        qrA[c].u[4] = qb.x; qrA[c].u[5] = qb.y; qrA[c].u[6] = qb.z; qrA[c].u[7] = qb.w;
        uint4 ia = *(const uint4*)(qib + kb0);
        uint4 ib = *(const uint4*)(qib + kb0 + 8);
        qiA[c].u[0] = ia.x; qiA[c].u[1] = ia.y; qiA[c].u[2] = ia.z; qiA[c].u[3] = ia.w;
        qiA[c].u[4] = ib.x; qiA[c].u[5] = ib.y; qiA[c].u[6] = ib.z; qiA[c].u[7] = ib.w;
    }

    // ---- fill only DEAD slots of the 10-slot window with masked value ----
    const float fill = -NEGBIG * temp;
    {
        float4 f4 = {fill, fill, fill, fill};
        int dead = 9 - (qt < 8 ? qt : 8);     // slots never written by compute
        int r2 = lane >> 1;                   // row 0..15
        int cq = (lane & 1) * 2;              // float4 slot within 16 cols
        for (int t = 0; t < dead; ++t) {
            float4* dst = (float4*)(scw + r2 * SCS + t * 16) + cq;
            dst[0] = f4;
            dst[1] = f4;
        }
    }

    // ---- scores: Q @ K^T over live key tiles ----
    const int kt0 = (qt - 8 < 0) ? 0 : qt - 8;
    for (int kt = kt0; kt <= qt; ++kt) {
        const __bf16* krb = kr + (size_t)(b * SEQ + kt * 16 + n15) * KD;
        const __bf16* kib = ki + (size_t)(b * SEQ + kt * 16 + n15) * KD;
        v8f acc;
#pragma unroll
        for (int r = 0; r < 8; ++r) acc[r] = 0.0f;
#pragma unroll
        for (int c = 0; c < 2; ++c) {
            int base = c * 32 + hh * 16;   // B layout: lane = key col, K = feat
            VB bk, bi;
            uint4 k0 = *(const uint4*)(krb + base);
            uint4 k1 = *(const uint4*)(krb + base + 8);
            bk.u[0] = k0.x; bk.u[1] = k0.y; bk.u[2] = k0.z; bk.u[3] = k0.w;
            bk.u[4] = k1.x; bk.u[5] = k1.y; bk.u[6] = k1.z; bk.u[7] = k1.w;
            uint4 i0 = *(const uint4*)(kib + base);
            uint4 i1 = *(const uint4*)(kib + base + 8);
            bi.u[0] = i0.x; bi.u[1] = i0.y; bi.u[2] = i0.z; bi.u[3] = i0.w;
            bi.u[4] = i1.x; bi.u[5] = i1.y; bi.u[6] = i1.z; bi.u[7] = i1.w;
            acc = wmma_bf16(qrA[c].v, bk.v, acc);
            acc = wmma_bf16(qiA[c].v, bi.v, acc);   // qi pre-negated
        }
        const int relt = kt - (qt - 9);   // slot 1..9 of 10-slot window
#pragma unroll
        for (int r = 0; r < 8; ++r) {
            int m = r + (hh << 3);
            bool masked = (kt == qt && n15 > m) ||      // causal edge
                          (kt == qt - 8 && n15 < m);    // band lower edge
            float s = masked ? -NEGBIG : acc[r] * QSCALE;
            scw[m * SCS + relt * 16 + n15] = s * temp;
        }
    }

    // ---- softmax: 2 lanes/row, float4 LDS sweeps + shfl_xor(16) ----
    {
        float4* rowv = (float4*)(scw + n15 * SCS + hh * 80);
        float mx = -3.0e38f;
        for (int i = 0; i < 20; ++i) {
            float4 t = rowv[i];
            mx = fmaxf(mx, fmaxf(fmaxf(t.x, t.y), fmaxf(t.z, t.w)));
        }
        mx = fmaxf(mx, __shfl_xor(mx, 16));
        float sum = 0.0f;
        for (int i = 0; i < 20; ++i) {
            float4 t = rowv[i];
            t.x = __expf(t.x - mx); t.y = __expf(t.y - mx);
            t.z = __expf(t.z - mx); t.w = __expf(t.w - mx);
            rowv[i] = t;
            sum += (t.x + t.y) + (t.z + t.w);
        }
        sum += __shfl_xor(sum, 16);
        float inv = 1.0f / sum;
        uint2* pv = (uint2*)(pbw + n15 * SCS + hh * 80);
        for (int i = 0; i < 20; ++i) {
            float4 t = rowv[i];
            uint2 w;
            w.x = pack2(t.x * inv, t.y * inv);
            w.y = pack2(t.z * inv, t.w * inv);
            pv[i] = w;
        }
    }

    // ---- P @ V over 5 chunks of K=32 keys ----
    v8f oacc[4];
#pragma unroll
    for (int nt = 0; nt < 4; ++nt)
#pragma unroll
        for (int r = 0; r < 8; ++r) oacc[nt][r] = 0.0f;

    const int kbase0 = (qt - 9) * 16;
    const __bf16* vTb = vT + (size_t)b * KD * SEQ;
    for (int c = 0; c < 5; ++c) {
        VB pA;   // probs in A layout from LDS (b64 reads, 8B aligned)
        {
            int off = n15 * SCS + c * 32 + hh * 16;
            uint2 p0 = *(const uint2*)(pbw + off);
            uint2 p1 = *(const uint2*)(pbw + off + 4);
            uint2 p2 = *(const uint2*)(pbw + off + 8);
            uint2 p3 = *(const uint2*)(pbw + off + 12);
            pA.u[0] = p0.x; pA.u[1] = p0.y; pA.u[2] = p1.x; pA.u[3] = p1.y;
            pA.u[4] = p2.x; pA.u[5] = p2.y; pA.u[6] = p3.x; pA.u[7] = p3.y;
        }
        int base = kbase0 + c * 32 + (hh << 4);   // key index of K pair j=0
#pragma unroll
        for (int nt = 0; nt < 4; ++nt) {
            int d = nt * 16 + n15;
            const __bf16* vrow = vTb + (size_t)d * SEQ;
            VB bV;   // B layout: lane = output-dim col, K pairs along key
            if (base >= 0) {
                uint4 a0 = *(const uint4*)(vrow + base);
                uint4 a1 = *(const uint4*)(vrow + base + 8);
                bV.u[0] = a0.x; bV.u[1] = a0.y; bV.u[2] = a0.z; bV.u[3] = a0.w;
                bV.u[4] = a1.x; bV.u[5] = a1.y; bV.u[6] = a1.z; bV.u[7] = a1.w;
            } else {
#pragma unroll
                for (int j = 0; j < 8; ++j) {
                    int kg = base + 2 * j;
                    int kgc = kg < 0 ? 0 : kg;    // probs are 0 there anyway
                    bV.u[j] = *(const unsigned*)(vrow + kgc);
                }
            }
            oacc[nt] = wmma_bf16(pA.v, bV.v, oacc[nt]);
        }
    }

    // ---- write output (C layout -> [b, s, d] f32) ----
#pragma unroll
    for (int nt = 0; nt < 4; ++nt)
#pragma unroll
        for (int r = 0; r < 8; ++r) {
            int m = r + (hh << 3);
            out[(size_t)(b * SEQ + qt * 16 + m) * KD + nt * 16 + n15] =
                oacc[nt][r];
        }
}

// ---------------------------------------------------------------------------
extern "C" void kernel_launch(void* const* d_in, const int* in_sizes, int n_in,
                              void* d_out, int out_size, void* d_ws, size_t ws_size,
                              hipStream_t stream) {
    const float* x    = (const float*)d_in[0];
    const float* Wqr  = (const float*)d_in[1];
    const float* bqr  = (const float*)d_in[2];
    const float* Wqi  = (const float*)d_in[3];
    const float* bqi  = (const float*)d_in[4];
    const float* Wkr  = (const float*)d_in[5];
    const float* bkr  = (const float*)d_in[6];
    const float* Wki  = (const float*)d_in[7];
    const float* bki  = (const float*)d_in[8];
    const float* Wv   = (const float*)d_in[9];
    const float* bv   = (const float*)d_in[10];
    const float* pqr  = (const float*)d_in[11];
    const float* pqi  = (const float*)d_in[12];
    const float* pkr  = (const float*)d_in[13];
    const float* pki  = (const float*)d_in[14];
    const float* temp = (const float*)d_in[15];

    __bf16* wsb = (__bf16*)d_ws;
    const size_t N = (size_t)BATCH * SEQ * KD;   // 1M elements (2 MB bf16)
    __bf16* qr  = wsb + 0 * N;
    __bf16* qiN = wsb + 1 * N;
    __bf16* kr  = wsb + 2 * N;
    __bf16* ki  = wsb + 3 * N;
    __bf16* vT  = wsb + 4 * N;   // transposed [b][d][s]

    proj_kernel<<<dim3(256, 5), 128, 0, stream>>>(
        x, Wqr, bqr, Wqi, bqi, Wkr, bkr, Wki, bki, Wv, bv,
        pqr, pqi, pkr, pki, qr, qiN, kr, ki, vT);

    attn_kernel<<<dim3(256), 128, 0, stream>>>(
        qr, qiN, kr, ki, vT, temp, (float*)d_out);
}
